// LinearAttention_5411658793245
// MI455X (gfx1250) — matrix operand order repaired and checked by reading, not compiled, not run on hardware
//
#include <hip/hip_runtime.h>
#include <hip/hip_bf16.h>

// Problem constants (from reference)
constexpr int BATCH = 2;
constexpr int SEQ   = 2048;
constexpr int HIDN  = 1024;
constexpr int NHEAD = 16;
constexpr int NKVH  = 4;
constexpr int HDIM  = 64;
constexpr int NREPC = NHEAD / NKVH;

typedef __attribute__((ext_vector_type(16))) _Float16 v16h;
typedef __attribute__((ext_vector_type(8)))  _Float16 v8h;
typedef __attribute__((ext_vector_type(8)))  float    v8f;

// ---------------------------------------------------------------------------
// Fragment loaders per CDNA5 ISA wave32 layouts (CDNA5 ISA §7.12.2)
// A 16x32 f16: lane L holds row M=L%16; K chunk of 8 at K=(L<16?0:8), second
// chunk of 8 at +16.  B 32x16 (given row-major [N][K], i.e. B^T): lane L holds
// column N=L%16, same K layout as A: chunk of 8 at K=(L<16?0:8), second at +16.
// ---------------------------------------------------------------------------
__device__ __forceinline__ v16h load_fragA(const _Float16* p, int ld) {
  const int lane = threadIdx.x & 31;
  const _Float16* q = p + (size_t)(lane & 15) * ld + ((lane >> 4) << 3);
  v8h lo = *(const v8h*)(q);
  v8h hi = *(const v8h*)(q + 16);
  v16h f;
#pragma unroll
  for (int i = 0; i < 8; ++i) { f[i] = lo[i]; f[i + 8] = hi[i]; }
  return f;
}

__device__ __forceinline__ v16h load_fragB(const _Float16* p, int ld) {
  const int lane = threadIdx.x & 31;
  const _Float16* q = p + (size_t)(lane & 15) * ld + ((lane >> 4) << 3);
  v8h lo = *(const v8h*)(q);
  v8h hi = *(const v8h*)(q + 16);
  v16h f;
#pragma unroll
  for (int i = 0; i < 8; ++i) { f[i] = lo[i]; f[i + 8] = hi[i]; }
  return f;
}

#define WMMA_F16(Aa, Bb, Cc) \
  __builtin_amdgcn_wmma_f32_16x16x32_f16(false, (Aa), false, (Bb), (short)0, (Cc), false, false)

// ---------------------------------------------------------------------------
// Generic batched GEMM: C[M,N](f32,row-major) = A[M,K](f16) * Bt[N,K](f16)^T
// blockDim = 256 (8 waves); one wave computes a 32x64 tile (2 A-frags x 4
// B-frags -> 8 WMMAs per K-step), with register double-buffering so next
// K-step fragment loads overlap the current WMMAs. blockIdx.y = batch.
// Requires M%32==0, N%64==0, K%32==0.
// ---------------------------------------------------------------------------
__global__ void __launch_bounds__(256)
gemm_f16_bt(const _Float16* __restrict__ A, const _Float16* __restrict__ Bt,
            float* __restrict__ C, int M, int N, int K,
            long long sA, long long sB, long long sC) {
  const int wave  = blockIdx.x * 8 + (threadIdx.x >> 5);
  const int nG    = N >> 6;
  const int tiles = (M >> 5) * nG;
  if (wave >= tiles) return;
  const int mT = wave / nG;           // 32-row tile index
  const int nB = (wave % nG) << 6;    // 64-col group base

  const _Float16* Ab = A  + (long long)blockIdx.y * sA + (size_t)mT * 32 * K;
  const _Float16* Bb = Bt + (long long)blockIdx.y * sB + (size_t)nB * K;
  float*          Cb = C  + (long long)blockIdx.y * sC;

  v8f c00 = {}, c01 = {}, c02 = {}, c03 = {};
  v8f c10 = {}, c11 = {}, c12 = {}, c13 = {};

  // prologue: fragments for k0 = 0
  v16h a0 = load_fragA(Ab, K);
  v16h a1 = load_fragA(Ab + (size_t)16 * K, K);
  v16h b0 = load_fragB(Bb + (size_t) 0 * K, K);
  v16h b1 = load_fragB(Bb + (size_t)16 * K, K);
  v16h b2 = load_fragB(Bb + (size_t)32 * K, K);
  v16h b3 = load_fragB(Bb + (size_t)48 * K, K);

  for (int k0 = 32; k0 < K; k0 += 32) {
    __builtin_prefetch(Ab + k0 + 32, 0, 3);
    // issue next K-step loads first so they overlap the 8 WMMAs below
    v16h na0 = load_fragA(Ab + k0, K);
    v16h na1 = load_fragA(Ab + (size_t)16 * K + k0, K);
    v16h nb0 = load_fragB(Bb + (size_t) 0 * K + k0, K);
    v16h nb1 = load_fragB(Bb + (size_t)16 * K + k0, K);
    v16h nb2 = load_fragB(Bb + (size_t)32 * K + k0, K);
    v16h nb3 = load_fragB(Bb + (size_t)48 * K + k0, K);

    c00 = WMMA_F16(a0, b0, c00);
    c01 = WMMA_F16(a0, b1, c01);
    c02 = WMMA_F16(a0, b2, c02);
    c03 = WMMA_F16(a0, b3, c03);
    c10 = WMMA_F16(a1, b0, c10);
    c11 = WMMA_F16(a1, b1, c11);
    c12 = WMMA_F16(a1, b2, c12);
    c13 = WMMA_F16(a1, b3, c13);

    a0 = na0; a1 = na1;
    b0 = nb0; b1 = nb1; b2 = nb2; b3 = nb3;
  }
  // epilogue K-step
  c00 = WMMA_F16(a0, b0, c00);
  c01 = WMMA_F16(a0, b1, c01);
  c02 = WMMA_F16(a0, b2, c02);
  c03 = WMMA_F16(a0, b3, c03);
  c10 = WMMA_F16(a1, b0, c10);
  c11 = WMMA_F16(a1, b1, c11);
  c12 = WMMA_F16(a1, b2, c12);
  c13 = WMMA_F16(a1, b3, c13);

  // C/D layout: VGPR r -> row r (lanes 0-15) / row 8+r (lanes 16-31), col = lane%16
  const int lane = threadIdx.x & 31;
  const int col  = lane & 15;
  const int rb   = (lane >> 4) << 3;
  float* Crow0 = Cb + (size_t)(mT * 32 + rb) * N + nB + col;
  float* Crow1 = Cb + (size_t)(mT * 32 + 16 + rb) * N + nB + col;
#pragma unroll
  for (int r = 0; r < 8; ++r) {
    Crow0[(size_t)r * N +  0] = c00[r];
    Crow0[(size_t)r * N + 16] = c01[r];
    Crow0[(size_t)r * N + 32] = c02[r];
    Crow0[(size_t)r * N + 48] = c03[r];
    Crow1[(size_t)r * N +  0] = c10[r];
    Crow1[(size_t)r * N + 16] = c11[r];
    Crow1[(size_t)r * N + 32] = c12[r];
    Crow1[(size_t)r * N + 48] = c13[r];
  }
}

// ---------------------------------------------------------------------------
// Elementwise helpers
// ---------------------------------------------------------------------------
__global__ void cvt_f32_f16(const float* __restrict__ in, _Float16* __restrict__ out, int n) {
  int i = blockIdx.x * blockDim.x + threadIdx.x;
  if (i < n) out[i] = (_Float16)in[i];
}

// in: [K][N] f32 row-major  ->  out: [N][K] f16 row-major (transposed)
__global__ void tcvt_f32_f16(const float* __restrict__ in, _Float16* __restrict__ out,
                             int K, int N) {
  int i = blockIdx.x * blockDim.x + threadIdx.x;
  if (i >= K * N) return;
  int n = i / K, k = i - n * K;
  out[i] = (_Float16)in[(size_t)k * N + n];
}

// RoPE + elu+1 feature map. One block per (b,s) token, 256 threads.
// Outputs: qf  f16 [b][h][s][d]           (A-matrix for q@State GEMM)
//          kfT f16 [b][kv][d][s] (masked) (B-matrix rows for state GEMM)
//          vT  f16 [b][kv][d][s]          (A-matrix rows for state GEMM)
__global__ void __launch_bounds__(256)
rope_feat(const float* __restrict__ qb, const float* __restrict__ kb,
          const float* __restrict__ vb, const float* __restrict__ cosb,
          const float* __restrict__ sinb, const int* __restrict__ mask,
          _Float16* __restrict__ qf, _Float16* __restrict__ kfT,
          _Float16* __restrict__ vT) {
  const int bs = blockIdx.x;
  const int b  = bs / SEQ, s = bs - b * SEQ;
  const int t  = threadIdx.x;
  const float* cp = cosb + (size_t)s * HDIM;
  const float* sp = sinb + (size_t)s * HDIM;
  const float msc = (mask[(size_t)b * SEQ + s] != 0) ? 1.0f : 0.0f;

  const float* qrow = qb + (size_t)bs * (NHEAD * HDIM);
#pragma unroll
  for (int j = 0; j < 4; ++j) {
    int idx = t + j * 256;
    int h = idx >> 6, d = idx & 63;
    float x = qrow[idx];
    float other = (d < 32) ? -qrow[(h << 6) + d + 32] : qrow[(h << 6) + d - 32];
    float r = x * cp[d] + other * sp[d];
    float f = (r > 0.f) ? (r + 1.f) : __expf(r);   // elu(r)+1
    qf[((size_t)(b * NHEAD + h) * SEQ + s) * HDIM + d] = (_Float16)f;
  }
  {
    int idx = t;                      // 256 = NKVH*HDIM
    int kv = idx >> 6, d = idx & 63;
    const float* krow = kb + (size_t)bs * (NKVH * HDIM);
    float x = krow[idx];
    float other = (d < 32) ? -krow[(kv << 6) + d + 32] : krow[(kv << 6) + d - 32];
    float r = x * cp[d] + other * sp[d];
    float f = ((r > 0.f) ? (r + 1.f) : __expf(r)) * msc;
    size_t o = ((size_t)(b * NKVH + kv) * HDIM + d) * SEQ + s;
    kfT[o] = (_Float16)f;
    vT[o]  = (_Float16)vb[(size_t)bs * (NKVH * HDIM) + idx];
  }
}

// ksum[row] = sum_s kfT[row][s];  one block (256 thr) per row, LDS reduction.
__global__ void __launch_bounds__(256)
ksum_kernel(const _Float16* __restrict__ kfT, float* __restrict__ ksum) {
  __shared__ float red[256];
  const _Float16* p = kfT + (size_t)blockIdx.x * SEQ;
  float acc = 0.f;
  for (int i = threadIdx.x; i < SEQ; i += 256) acc += (float)p[i];
  red[threadIdx.x] = acc;
  __syncthreads();
  for (int off = 128; off > 0; off >>= 1) {
    if (threadIdx.x < off) red[threadIdx.x] += red[threadIdx.x + off];
    __syncthreads();
  }
  if (threadIdx.x == 0) ksum[blockIdx.x] = red[0];
}

// Replicate per-kv-head StateT (f32 64x64) to per-q-head f16 copies (GQA).
__global__ void expand_state(const float* __restrict__ st, _Float16* __restrict__ sth) {
  int i = blockIdx.x * blockDim.x + threadIdx.x;     // BATCH*NHEAD*4096
  if (i >= BATCH * NHEAD * HDIM * HDIM) return;
  int e  = i & 4095;
  int bh = i >> 12;
  int b = bh / NHEAD, h = bh - b * NHEAD;
  sth[i] = (_Float16)st[(((size_t)(b * NKVH + h / NREPC)) << 12) + e];
}

// denom[b,h,s] = max(q_feat . ksum[b, h/NREP], 1e-6)
__global__ void denom_kernel(const _Float16* __restrict__ qf, const float* __restrict__ ksum,
                             float* __restrict__ denom) {
  int i = blockIdx.x * blockDim.x + threadIdx.x;     // BATCH*NHEAD*SEQ
  if (i >= BATCH * NHEAD * SEQ) return;
  int bh = i / SEQ;
  int b = bh / NHEAD, h = bh - b * NHEAD;
  const _Float16* q = qf + (size_t)i * HDIM;
  const float* ks = ksum + (size_t)(b * NKVH + h / NREPC) * HDIM;
  float acc = 0.f;
#pragma unroll
  for (int d = 0; d < HDIM; ++d) acc += (float)q[d] * ks[d];
  denom[i] = fmaxf(acc, 1e-6f);
}

// attn[(b*SEQ+s)*1024 + h*64+d] = headout[b,h,s,d] / denom[b,h,s]   (f16)
__global__ void epilogue(const float* __restrict__ headout, const float* __restrict__ denom,
                         _Float16* __restrict__ attn) {
  int i = blockIdx.x * blockDim.x + threadIdx.x;     // BATCH*NHEAD*SEQ*HDIM
  if (i >= BATCH * NHEAD * SEQ * HDIM) return;
  int d   = i & 63;
  int bhs = i >> 6;
  int s  = bhs % SEQ;
  int bh = bhs / SEQ;
  int b = bh / NHEAD, h = bh - b * NHEAD;
  float val = headout[i] / denom[bhs];
  attn[((size_t)(b * SEQ + s) * (NHEAD * HDIM)) + h * HDIM + d] = (_Float16)val;
}

// ---------------------------------------------------------------------------
extern "C" void kernel_launch(void* const* d_in, const int* in_sizes, int n_in,
                              void* d_out, int out_size, void* d_ws, size_t ws_size,
                              hipStream_t stream) {
  const float* x    = (const float*)d_in[0];
  const float* cosb = (const float*)d_in[1];
  const float* sinb = (const float*)d_in[2];
  const int*   mask = (const int*)d_in[3];
  const float* Wq   = (const float*)d_in[4];
  const float* Wk   = (const float*)d_in[5];
  const float* Wv   = (const float*)d_in[6];
  const float* Wo   = (const float*)d_in[7];
  float* out = (float*)d_out;

  char* ws = (char*)d_ws;
  size_t off = 0;
  auto carve = [&](size_t bytes) -> void* {
    off = (off + 255) & ~(size_t)255;
    void* p = ws + off;
    off += bytes;
    return p;
  };

  const size_t M = (size_t)BATCH * SEQ;                                   // 4096
  _Float16* xh    = (_Float16*)carve(M * HIDN * 2);                       // x  f16
  _Float16* WqT   = (_Float16*)carve((size_t)HIDN * (NHEAD * HDIM) * 2);  // [N][K]
  _Float16* WkT   = (_Float16*)carve((size_t)HIDN * (NKVH * HDIM) * 2);
  _Float16* WvT   = (_Float16*)carve((size_t)HIDN * (NKVH * HDIM) * 2);
  _Float16* WoT   = (_Float16*)carve((size_t)(NHEAD * HDIM) * HIDN * 2);
  float*    qbuf  = (float*)carve(M * (NHEAD * HDIM) * 4);
  float*    kbuf  = (float*)carve(M * (NKVH * HDIM) * 4);
  float*    vbuf  = (float*)carve(M * (NKVH * HDIM) * 4);
  _Float16* qf    = (_Float16*)carve(M * (NHEAD * HDIM) * 2);             // [b,h,s,d]
  _Float16* kfT   = (_Float16*)carve((size_t)BATCH * NKVH * HDIM * SEQ * 2);
  _Float16* vT    = (_Float16*)carve((size_t)BATCH * NKVH * HDIM * SEQ * 2);
  float*    ksumb = (float*)carve((size_t)BATCH * NKVH * HDIM * 4);
  float*    stT   = (float*)carve((size_t)BATCH * NKVH * HDIM * HDIM * 4);
  _Float16* stTh  = (_Float16*)carve((size_t)BATCH * NHEAD * HDIM * HDIM * 2);
  float*    hout  = (float*)carve((size_t)BATCH * NHEAD * SEQ * HDIM * 4);
  float*    denb  = (float*)carve((size_t)BATCH * NHEAD * SEQ * 4);
  _Float16* attn  = (_Float16*)carve(M * (NHEAD * HDIM) * 2);

  auto launch_gemm = [&](const _Float16* A, const _Float16* Bt, float* C,
                         int m, int n, int k, int batch,
                         long long sA, long long sB, long long sC) {
    int tiles = (m / 32) * (n / 64);
    dim3 grid((tiles + 7) / 8, batch);
    gemm_f16_bt<<<grid, 256, 0, stream>>>(A, Bt, C, m, n, k, sA, sB, sC);
  };

  // 1) convert x to f16; transpose-convert weights to [N][K] f16
  {
    int n = (int)(M * HIDN);
    cvt_f32_f16<<<(n + 255) / 256, 256, 0, stream>>>(x, xh, n);
  }
  tcvt_f32_f16<<<(HIDN * NHEAD * HDIM + 255) / 256, 256, 0, stream>>>(Wq, WqT, HIDN, NHEAD * HDIM);
  tcvt_f32_f16<<<(HIDN * NKVH * HDIM + 255) / 256, 256, 0, stream>>>(Wk, WkT, HIDN, NKVH * HDIM);
  tcvt_f32_f16<<<(HIDN * NKVH * HDIM + 255) / 256, 256, 0, stream>>>(Wv, WvT, HIDN, NKVH * HDIM);
  tcvt_f32_f16<<<(NHEAD * HDIM * HIDN + 255) / 256, 256, 0, stream>>>(Wo, WoT, NHEAD * HDIM, HIDN);

  // 2) q/k/v projections (WMMA)
  launch_gemm(xh, WqT, qbuf, (int)M, NHEAD * HDIM, HIDN, 1, 0, 0, 0);
  launch_gemm(xh, WkT, kbuf, (int)M, NKVH * HDIM, HIDN, 1, 0, 0, 0);
  launch_gemm(xh, WvT, vbuf, (int)M, NKVH * HDIM, HIDN, 1, 0, 0, 0);

  // 3) RoPE + elu+1 feature map (+ mask on k), transposed k/v layouts
  rope_feat<<<BATCH * SEQ, 256, 0, stream>>>(qbuf, kbuf, vbuf, cosb, sinb, mask, qf, kfT, vT);

  // 4) StateT[b,kv][m][n] = sum_s v[s][m] * k_feat[s][n]   (batched WMMA 64x64x2048)
  launch_gemm(vT, kfT, stT, HDIM, HDIM, SEQ, BATCH * NKVH,
              (long long)HDIM * SEQ, (long long)HDIM * SEQ, (long long)HDIM * HDIM);

  // 5) ksum reduction (LDS)
  ksum_kernel<<<BATCH * NKVH * HDIM, 256, 0, stream>>>(kfT, ksumb);

  // 6) expand state per q-head (GQA) -> f16
  {
    int n = BATCH * NHEAD * HDIM * HDIM;
    expand_state<<<(n + 255) / 256, 256, 0, stream>>>(stT, stTh);
  }

  // 7) headout[b,h] = q_feat[b,h] (SEQ x 64) @ State  (batched WMMA 2048x64x64)
  launch_gemm(qf, stTh, hout, SEQ, HDIM, HDIM, BATCH * NHEAD,
              (long long)SEQ * HDIM, (long long)HDIM * HDIM, (long long)SEQ * HDIM);

  // 8) denom + normalize + repack token-major f16
  {
    int n = BATCH * NHEAD * SEQ;
    denom_kernel<<<(n + 255) / 256, 256, 0, stream>>>(qf, ksumb, denb);
  }
  {
    int n = BATCH * NHEAD * SEQ * HDIM;
    epilogue<<<(n + 255) / 256, 256, 0, stream>>>(hout, denb, attn);
  }

  // 9) final projection: out = attn @ Wo  (WMMA, f32 out)
  launch_gemm(attn, WoT, out, (int)M, HIDN, NHEAD * HDIM, 1, 0, 0, 0);
}
